// XToQ_10969346474266
// MI455X (gfx1250) — compile-verified, matrix operand order checked by reading, not run on hardware
//
#include <hip/hip_runtime.h>
#include <hip/hip_bf16.h>

// Problem constants (from reference): S=1024, B=32, H=1024, T=512
#define S_DIM 1024
#define B_DIM 32
#define H_DIM 1024
#define T_DIM 512

typedef __attribute__((ext_vector_type(16))) __bf16 v16bf;
typedef __attribute__((ext_vector_type(8)))  __bf16 v8bf;
typedef __attribute__((ext_vector_type(4)))  __bf16 v4bf;
typedef __attribute__((ext_vector_type(8)))  float  v8f;

__device__ __forceinline__ v8f wmma_bf16(v16bf a, v16bf b, v8f c) {
  // D(16x16 f32) = A(16x32 bf16) * B(32x16 bf16) + C
  return __builtin_amdgcn_wmma_f32_16x16x32_bf16(false, a, false, b, (short)0, c,
                                                 false, false);
}

__device__ __forceinline__ v16bf load16(const __bf16* __restrict__ p) {
  return *(const v16bf*)p;
}

// CDNA5 async global->LDS copy (GLOBAL_LOAD_ASYNC_TO_LDS_B128, ASYNCcnt).
// lds_addr: wave-relative LDS byte address; gaddr: 64-bit global address.
__device__ __forceinline__ void async_copy_b128(unsigned lds_addr,
                                                unsigned long long gaddr) {
  asm volatile("global_load_async_to_lds_b128 %0, %1, off"
               :: "v"(lds_addr), "v"(gaddr) : "memory");
}
__device__ __forceinline__ void wait_asynccnt0() {
  asm volatile("s_wait_asynccnt 0" ::: "memory");
}
__device__ __forceinline__ unsigned lds_off(const void* p) {
  return (unsigned)(size_t)p;  // flat LDS pointer: low 32 bits = LDS offset
}

// ------------------------------------------------------------ K0a: f32 -> bf16
__global__ __launch_bounds__(256) void cvt_f32_bf16_kernel(
    const float* __restrict__ in, __bf16* __restrict__ out, int n4) {
  int i = blockIdx.x * blockDim.x + threadIdx.x;
  if (i < n4) {
    const float4 f = ((const float4*)in)[i];
    v4bf o;
    o[0] = (__bf16)f.x; o[1] = (__bf16)f.y; o[2] = (__bf16)f.z; o[3] = (__bf16)f.w;
    ((v4bf*)out)[i] = o;
  }
}

// ---------------------------- K0b: hidden [S,B,H] f32 -> [B,S,H] bf16 transpose
// One 256-thread block handles one (s,b) row of H=1024 floats: coalesced both ways.
__global__ __launch_bounds__(256) void hidden_to_bf16_kernel(
    const float* __restrict__ hidden, __bf16* __restrict__ out) {
  const int sb = blockIdx.x;               // = s*B + b
  const int s = sb / B_DIM, b = sb % B_DIM;
  const float4 f = ((const float4*)(hidden + (size_t)sb * H_DIM))[threadIdx.x];
  v4bf o;
  o[0] = (__bf16)f.x; o[1] = (__bf16)f.y; o[2] = (__bf16)f.z; o[3] = (__bf16)f.w;
  ((v4bf*)(out + ((size_t)b * S_DIM + s) * H_DIM))[threadIdx.x] = o;
}

// ------------------------------------------------- K1: fused K/V projection GEMM
// k[b,s,n] = sum_h hid[b,s,h]*K_w[n,h] + K_b[n] -> kout  [B*S, H] (bf16)
// v[b,s,n] = sum_h hid[b,s,h]*V_w[n,h] + V_b[n] -> vtout [B, H, S] (bf16, transposed)
// All-bf16 inputs; one wave computes a 32x32 tile of both K and V (A shared).
__global__ __launch_bounds__(256) void kv_proj_kernel(
    const __bf16* __restrict__ hid,   // [B*S, H]
    const __bf16* __restrict__ Kw,    // [H, H] bf16
    const float* __restrict__ Kb,
    const __bf16* __restrict__ Vw,    // [H, H] bf16
    const float* __restrict__ Vb,
    __bf16* __restrict__ kout, __bf16* __restrict__ vtout) {
  const int lane = threadIdx.x & 31;
  const int wave = threadIdx.x >> 5;
  const int waveId = blockIdx.x * 8 + wave;
  const int NT = H_DIM / 32;
  const int nt = waveId % NT;
  const int mt = waveId / NT;
  const int r0 = mt * 32;                  // global row = b*S + s
  const int b  = r0 / S_DIM;
  const int s0 = r0 % S_DIM;
  const int n0 = nt * 32;
  const int lm = lane & 15, lh = lane >> 4;

  const __bf16* a0p = hid + (size_t)(r0 + lm) * H_DIM;
  const __bf16* a1p = hid + (size_t)(r0 + 16 + lm) * H_DIM;
  const __bf16* bk0p = Kw + (size_t)(n0 + lm) * H_DIM;
  const __bf16* bk1p = Kw + (size_t)(n0 + 16 + lm) * H_DIM;
  const __bf16* bv0p = Vw + (size_t)(n0 + lm) * H_DIM;
  const __bf16* bv1p = Vw + (size_t)(n0 + 16 + lm) * H_DIM;

  v8f cK[4] = {}; v8f cV[4] = {};
  for (int k0 = 0; k0 < H_DIM; k0 += 32) {
    const int kk = k0 + lh * 16;
    if (k0 + 32 < H_DIM) {
      __builtin_prefetch(a0p + kk + 32, 0, 1);
      __builtin_prefetch(bk0p + kk + 32, 0, 1);
      __builtin_prefetch(bv0p + kk + 32, 0, 1);
    }
    v16bf a0  = load16(a0p + kk);
    v16bf a1  = load16(a1p + kk);
    v16bf bk0 = load16(bk0p + kk);
    v16bf bk1 = load16(bk1p + kk);
    v16bf bv0 = load16(bv0p + kk);
    v16bf bv1 = load16(bv1p + kk);
    cK[0] = wmma_bf16(a0, bk0, cK[0]);
    cK[1] = wmma_bf16(a0, bk1, cK[1]);
    cK[2] = wmma_bf16(a1, bk0, cK[2]);
    cK[3] = wmma_bf16(a1, bk1, cK[3]);
    cV[0] = wmma_bf16(a0, bv0, cV[0]);
    cV[1] = wmma_bf16(a0, bv1, cV[1]);
    cV[2] = wmma_bf16(a1, bv0, cV[2]);
    cV[3] = wmma_bf16(a1, bv1, cV[3]);
  }

  const float kb[2] = { Kb[n0 + lm], Kb[n0 + 16 + lm] };
  const float vb[2] = { Vb[n0 + lm], Vb[n0 + 16 + lm] };

#pragma unroll
  for (int f = 0; f < 4; ++f) {
    const int mh = (f >> 1) * 16;
    const int nh = (f & 1) * 16;
    const int n  = n0 + nh + lm;
#pragma unroll
    for (int i = 0; i < 8; ++i) {
      const int m = mh + i + 8 * lh;
      kout[(size_t)(r0 + m) * H_DIM + n] = (__bf16)(cK[f][i] + kb[f & 1]);
    }
    v8bf pv;
#pragma unroll
    for (int i = 0; i < 8; ++i) pv[i] = (__bf16)(cV[f][i] + vb[f & 1]);
    *(v8bf*)(vtout + ((size_t)b * H_DIM + n) * S_DIM + s0 + mh + 8 * lh) = pv;
  }
}

// -------------------- K2/K4: batched C[m,n] = sum_k A[m,k]*Bt[n,k], LDS pipelined
// Block: 256 thr = 8 waves (4M x 2N), block tile 128M x 64N, K-step 32.
// Double-buffered async global->LDS staging (ASYNCcnt), fragments via ds_load.
template <bool SIG>
__global__ __launch_bounds__(256) void gemm_abt_lds_kernel(
    const __bf16* __restrict__ A,   // [nb, M, K]
    const __bf16* __restrict__ Bt,  // [nb, N, K]
    float* __restrict__ C,          // [nb, M, N]
    int M, int N, int K) {
  __shared__ __attribute__((aligned(32))) __bf16 lA[2][128 * 32];  // 16 KB
  __shared__ __attribute__((aligned(32))) __bf16 lB[2][64 * 32];   //  8 KB

  const int tid  = threadIdx.x;
  const int lane = tid & 31;
  const int wave = tid >> 5;
  const int wm = wave >> 1, wn = wave & 1;
  const int NB  = N / 64;
  const int bm0 = (blockIdx.x / NB) * 128;
  const int bn0 = (blockIdx.x % NB) * 64;
  const int bidx = blockIdx.y;

  const __bf16* Ab = A  + (size_t)bidx * M * K;
  const __bf16* Bb = Bt + (size_t)bidx * N * K;
  float*        Cb = C  + (size_t)bidx * M * N;

  // Copy assignment: each LDS row is 32 bf16 = 64 B = four 16 B chunks.
  // A tile: 128 rows -> 512 chunks (2/thread). B tile: 64 rows -> 256 chunks.
  const int cr = tid >> 2;                 // row within tile
  const int cb = (tid & 3) * 16;           // byte offset within row
  const unsigned long long gA0 =
      (unsigned long long)(size_t)(Ab + (size_t)(bm0 + cr) * K) + cb;
  const unsigned long long gA1 =
      (unsigned long long)(size_t)(Ab + (size_t)(bm0 + cr + 64) * K) + cb;
  const unsigned long long gB0 =
      (unsigned long long)(size_t)(Bb + (size_t)(bn0 + cr) * K) + cb;
  const unsigned lA0 = lds_off(&lA[0][cr * 32]) + cb;
  const unsigned lA1 = lds_off(&lA[0][(cr + 64) * 32]) + cb;
  const unsigned lB0 = lds_off(&lB[0][cr * 32]) + cb;

  const int lm = lane & 15, lh = lane >> 4;
  const __bf16* fa0 = &lA[0][(wm * 32 + lm) * 32 + lh * 16];
  const __bf16* fa1 = fa0 + 16 * 32;
  const __bf16* fb0 = &lB[0][(wn * 32 + lm) * 32 + lh * 16];
  const __bf16* fb1 = fb0 + 16 * 32;

  // Prologue: stage K-slab 0 into buffer 0.
  {
    const unsigned long long ko = 0;
    async_copy_b128(lA0, gA0 + ko);
    async_copy_b128(lA1, gA1 + ko);
    async_copy_b128(lB0, gB0 + ko);
  }
  wait_asynccnt0();
  __syncthreads();

  v8f c0 = {}, c1 = {}, c2 = {}, c3 = {};
  for (int k0 = 0; k0 < K; k0 += 32) {
    const int  cur  = (k0 >> 5) & 1;
    const bool more = (k0 + 32) < K;
    if (more) {  // stage next slab into the other buffer while we compute
      const unsigned long long ko = (unsigned long long)(k0 + 32) * 2ull;
      const unsigned bo = (cur ^ 1) ? 8192u : 0u;   // lA buffer byte stride
      const unsigned bb = (cur ^ 1) ? 4096u : 0u;   // lB buffer byte stride
      async_copy_b128(lA0 + bo, gA0 + ko);
      async_copy_b128(lA1 + bo, gA1 + ko);
      async_copy_b128(lB0 + bb, gB0 + ko);
    }
    const int ea = cur * (128 * 32);
    const int eb = cur * (64 * 32);
    v16bf a0 = *(const v16bf*)(fa0 + ea);
    v16bf a1 = *(const v16bf*)(fa1 + ea);
    v16bf b0 = *(const v16bf*)(fb0 + eb);
    v16bf b1 = *(const v16bf*)(fb1 + eb);
    c0 = wmma_bf16(a0, b0, c0);
    c1 = wmma_bf16(a0, b1, c1);
    c2 = wmma_bf16(a1, b0, c2);
    c3 = wmma_bf16(a1, b1, c3);
    if (more) wait_asynccnt0();   // own async issues complete
    __syncthreads();              // all waves' copies done + reads drained
  }

  const int m0 = bm0 + wm * 32, n0 = bn0 + wn * 32;
  v8f acc[4] = { c0, c1, c2, c3 };
#pragma unroll
  for (int f = 0; f < 4; ++f) {
    const int mh = (f >> 1) * 16;
    const int nh = (f & 1) * 16;
#pragma unroll
    for (int i = 0; i < 8; ++i) {
      float x = acc[f][i];
      if (SIG) x = 1.0f / (1.0f + __expf(-x));
      Cb[(size_t)(m0 + mh + i + 8 * lh) * N + n0 + nh + lm] = x;
    }
  }
}

// ---------------------------------------- K3: row softmax over S, f32 -> bf16
__global__ __launch_bounds__(256) void softmax_rows_kernel(
    const float* __restrict__ scores, __bf16* __restrict__ attn) {
  const int row  = blockIdx.x;          // B*T rows
  const int tid  = threadIdx.x;         // 256 threads, 4 elems each (S=1024)
  const int lane = tid & 31, wave = tid >> 5;
  const float* p = scores + (size_t)row * S_DIM;
  __bf16*      o = attn   + (size_t)row * S_DIM;

  float v[4]; float m = -INFINITY;
#pragma unroll
  for (int i = 0; i < 4; ++i) { v[i] = p[tid + i * 256]; m = fmaxf(m, v[i]); }
#pragma unroll
  for (int off = 16; off > 0; off >>= 1) m = fmaxf(m, __shfl_xor(m, off, 32));

  __shared__ float smax[8], ssum[8];
  if (lane == 0) smax[wave] = m;
  __syncthreads();
  if (wave == 0) {
    float mm = (lane < 8) ? smax[lane] : -INFINITY;
#pragma unroll
    for (int off = 4; off > 0; off >>= 1) mm = fmaxf(mm, __shfl_xor(mm, off, 32));
    if (lane == 0) smax[0] = mm;
  }
  __syncthreads();
  m = smax[0];

  float s = 0.0f;
#pragma unroll
  for (int i = 0; i < 4; ++i) { v[i] = __expf(v[i] - m); s += v[i]; }
#pragma unroll
  for (int off = 16; off > 0; off >>= 1) s += __shfl_xor(s, off, 32);
  if (lane == 0) ssum[wave] = s;
  __syncthreads();
  if (wave == 0) {
    float ss = (lane < 8) ? ssum[lane] : 0.0f;
#pragma unroll
    for (int off = 4; off > 0; off >>= 1) ss += __shfl_xor(ss, off, 32);
    if (lane == 0) ssum[0] = ss;
  }
  __syncthreads();
  const float inv = 1.0f / ssum[0];
#pragma unroll
  for (int i = 0; i < 4; ++i) o[tid + i * 256] = (__bf16)(v[i] * inv);
}

// ------------------------------------------------------------------- launcher
extern "C" void kernel_launch(void* const* d_in, const int* in_sizes, int n_in,
                              void* d_out, int out_size, void* d_ws, size_t ws_size,
                              hipStream_t stream) {
  const float* hidden = (const float*)d_in[0];  // [S,B,H]
  const float* x      = (const float*)d_in[1];  // [B,T,H]
  // d_in[2] = problem_q (unused by reference)
  const float* Kw     = (const float*)d_in[3];
  const float* Kb     = (const float*)d_in[4];
  const float* Vw     = (const float*)d_in[5];
  const float* Vb     = (const float*)d_in[6];
  float* out = (float*)d_out;                   // [B,T,H] f32

  // Workspace (228 MB, with stream-ordered buffer reuse):
  //   [0,32M)    x_bf [B,T,H]      -> reused as attn [B,T,S] bf16 after K2
  //   [32M,96M)  hid_bf [B,S,H]    -> reused as scores [B,T,S] f32 after K1
  //   [96M,98M)  Kw_bf ; [98M,100M) Vw_bf
  //   [100M,164M) k_bf [B*S,H] ; [164M,228M) vt_bf [B,H,S]
  char* ws = (char*)d_ws;
  __bf16* x_bf   = (__bf16*)(ws);
  __bf16* attn   = (__bf16*)(ws);
  __bf16* hid_bf = (__bf16*)(ws + ((size_t)32  << 20));
  float*  scores = (float*) (ws + ((size_t)32  << 20));
  __bf16* Kw_bf  = (__bf16*)(ws + ((size_t)96  << 20));
  __bf16* Vw_bf  = (__bf16*)(ws + ((size_t)98  << 20));
  __bf16* k_bf   = (__bf16*)(ws + ((size_t)100 << 20));
  __bf16* vt_bf  = (__bf16*)(ws + ((size_t)164 << 20));

  // K0: one-shot conversions (bandwidth-bound, removes cvt from GEMM loops)
  {
    const int n4x = (B_DIM * T_DIM * H_DIM) / 4;
    cvt_f32_bf16_kernel<<<n4x / 256, 256, 0, stream>>>(x, x_bf, n4x);
    const int n4w = (H_DIM * H_DIM) / 4;
    cvt_f32_bf16_kernel<<<n4w / 256, 256, 0, stream>>>(Kw, Kw_bf, n4w);
    cvt_f32_bf16_kernel<<<n4w / 256, 256, 0, stream>>>(Vw, Vw_bf, n4w);
    hidden_to_bf16_kernel<<<S_DIM * B_DIM, 256, 0, stream>>>(hidden, hid_bf);
  }

  // K1: fused K/V projection: (B*S/32)*(H/32)=32768 wave tiles / 8 per block
  kv_proj_kernel<<<4096, 256, 0, stream>>>(hid_bf, Kw_bf, Kb, Vw_bf, Vb,
                                           k_bf, vt_bf);

  // K2: scores = x . k^T ; per batch (T/128)*(S/64) = 64 blocks
  gemm_abt_lds_kernel<false><<<dim3(64, B_DIM), 256, 0, stream>>>(
      x_bf, k_bf, scores, T_DIM, S_DIM, H_DIM);

  // K3: softmax over S for each of B*T rows
  softmax_rows_kernel<<<B_DIM * T_DIM, 256, 0, stream>>>(scores, attn);

  // K4: out = sigmoid(attn . v) ; v stored [B,H,S] -> same AxB^T pattern
  gemm_abt_lds_kernel<true><<<dim3(64, B_DIM), 256, 0, stream>>>(
      attn, vt_bf, out, T_DIM, H_DIM, S_DIM);
}
